// MoE_82420422410976
// MI455X (gfx1250) — matrix operand order repaired and checked by reading, not compiled, not run on hardware
//
#include <hip/hip_runtime.h>
#include <hip/hip_bf16.h>
#include <math.h>

typedef __attribute__((ext_vector_type(16))) __bf16 v16bf;
typedef __attribute__((ext_vector_type(8)))  float  v8f;
typedef __attribute__((ext_vector_type(4)))  int    v4i_t;

#define NUM_E   8
#define IND     1024
#define HDIM    2048
#define NTOK    16384            // BS*SEQ
#define NKTOT   32768            // NTOK*K
#define CAP     5120             // ceil(ceil(NK/E)*1.25)
#define NSLOT   (NUM_E*CAP)      // 40960
#define DUMMYS  NSLOT
#define HBLK    64
#define HCHUNK  (NKTOT/HBLK)     // 512
#define KP      40               // LDS K pitch (32 + 8 pad) -> conflict-free b128 reads

union Frag16 { v16bf v; uint4 q[2]; };

// ---------- CDNA5 async global->LDS staging (ASYNCcnt) with safe fallback ----------
#if defined(__HIP_DEVICE_COMPILE__) && __has_builtin(__builtin_amdgcn_global_load_async_to_lds_b128)
#define HAVE_ASYNC_LDS 1
typedef __attribute__((address_space(1))) v4i_t gas_v4i;   // prints as "__device__" in diags
typedef __attribute__((address_space(3))) v4i_t las_v4i;   // prints as "__shared__"
__device__ __forceinline__ void async_g2l_b128(const void* g, void* l) {
    __builtin_amdgcn_global_load_async_to_lds_b128((gas_v4i*)g, (las_v4i*)l, 0, 0);
}
#if __has_builtin(__builtin_amdgcn_s_wait_asynccnt)
#define WAIT_ASYNC0() __builtin_amdgcn_s_wait_asynccnt(0)
#else
#define WAIT_ASYNC0() asm volatile("s_wait_asynccnt 0x0" ::: "memory")
#endif
#else
#define HAVE_ASYNC_LDS 0
#define WAIT_ASYNC0()
#endif

// ---------------- f32 -> bf16 cast (float4 at a time) ----------------
__global__ __launch_bounds__(256) void cast_bf16_kernel(const float* __restrict__ in,
                                                        __bf16* __restrict__ out, int n4) {
    int i = blockIdx.x * 256 + threadIdx.x;
    if (i >= n4) return;
    float4 v = reinterpret_cast<const float4*>(in)[i];
    union { __bf16 h[4]; ushort4 u; } o;
    o.h[0] = (__bf16)v.x; o.h[1] = (__bf16)v.y; o.h[2] = (__bf16)v.z; o.h[3] = (__bf16)v.w;
    reinterpret_cast<ushort4*>(out)[i] = o.u;
}

__global__ __launch_bounds__(256) void init_slots_kernel(int* __restrict__ slot_token) {
    int i = blockIdx.x * 256 + threadIdx.x;
    if (i < NSLOT) slot_token[i] = -1;
}

// ---------------- routing: one wave32 per token ----------------
__global__ __launch_bounds__(256) void routing_kernel(
    const float* __restrict__ x, const float* __restrict__ gate_w,
    int* __restrict__ exp_ids, float* __restrict__ wts, float* __restrict__ probs)
{
    const int wave = threadIdx.x >> 5, lane = threadIdx.x & 31;
    const int n = blockIdx.x * 8 + wave;
    float part[NUM_E];
#pragma unroll
    for (int e = 0; e < NUM_E; ++e) part[e] = 0.f;
    const float* xr = x + (size_t)n * IND;
    for (int j = 0; j < IND / 32; ++j) {
        float xv = xr[lane + j * 32];
#pragma unroll
        for (int e = 0; e < NUM_E; ++e)
            part[e] += xv * gate_w[e * IND + lane + j * 32];
    }
#pragma unroll
    for (int e = 0; e < NUM_E; ++e) {
#pragma unroll
        for (int off = 16; off >= 1; off >>= 1)
            part[e] += __shfl_xor(part[e], off, 32);
    }
    if (lane == 0) {
        float mx = part[0];
#pragma unroll
        for (int e = 1; e < NUM_E; ++e) mx = fmaxf(mx, part[e]);
        float p[NUM_E], s = 0.f;
#pragma unroll
        for (int e = 0; e < NUM_E; ++e) { p[e] = __expf(part[e] - mx); s += p[e]; }
        float inv = 1.f / s;
#pragma unroll
        for (int e = 0; e < NUM_E; ++e) { p[e] *= inv; probs[(size_t)n * NUM_E + e] = p[e]; }
        // top-2, ties -> lower index (matches jax.lax.top_k)
        int i0 = 0; float v0 = p[0];
#pragma unroll
        for (int e = 1; e < NUM_E; ++e) if (p[e] > v0) { v0 = p[e]; i0 = e; }
        int i1 = 0; float v1 = -1.f;
#pragma unroll
        for (int e = 0; e < NUM_E; ++e) if (e != i0 && p[e] > v1) { v1 = p[e]; i1 = e; }
        float wsum = v0 + v1 + 1e-9f;
        exp_ids[n * 2 + 0] = i0; exp_ids[n * 2 + 1] = i1;
        wts[n * 2 + 0] = v0 / wsum; wts[n * 2 + 1] = v1 / wsum;
    }
}

// ---------------- per-block expert histograms (one wave per block) ----------------
__global__ __launch_bounds__(32) void hist_kernel(const int* __restrict__ exp_ids,
                                                  int* __restrict__ blockCounts) {
    const int b = blockIdx.x, lane = threadIdx.x;
    int cnt[NUM_E];
#pragma unroll
    for (int e = 0; e < NUM_E; ++e) cnt[e] = 0;
    for (int j = 0; j < HCHUNK / 32; ++j) {
        int eid = exp_ids[b * HCHUNK + j * 32 + lane];
#pragma unroll
        for (int e = 0; e < NUM_E; ++e) cnt[e] += (eid == e);
    }
#pragma unroll
    for (int e = 0; e < NUM_E; ++e) {
#pragma unroll
        for (int off = 16; off >= 1; off >>= 1) cnt[e] += __shfl_xor(cnt[e], off, 32);
    }
    if (lane == 0) {
#pragma unroll
        for (int e = 0; e < NUM_E; ++e) blockCounts[b * NUM_E + e] = cnt[e];
    }
}

// ---------------- serial scan (tiny: 64x8) ----------------
__global__ void scan_kernel(const int* __restrict__ blockCounts, int* __restrict__ blockStart,
                            int* __restrict__ expertStart, int* __restrict__ expertCount) {
    if (threadIdx.x != 0 || blockIdx.x != 0) return;
    int totals[NUM_E];
    for (int e = 0; e < NUM_E; ++e) {
        int s = 0;
        for (int b = 0; b < HBLK; ++b) s += blockCounts[b * NUM_E + e];
        totals[e] = s; expertCount[e] = s;
    }
    int st = 0;
    for (int e = 0; e < NUM_E; ++e) { expertStart[e] = st; st += totals[e]; }
    for (int e = 0; e < NUM_E; ++e) {
        int run = expertStart[e];
        for (int b = 0; b < HBLK; ++b) { blockStart[b * NUM_E + e] = run; run += blockCounts[b * NUM_E + e]; }
    }
}

// ---------------- stable scatter via wave ballots ----------------
__global__ __launch_bounds__(32) void scatter_kernel(
    const int* __restrict__ exp_ids, const int* __restrict__ blockStart,
    const int* __restrict__ expertStart, int* __restrict__ slot_of_i,
    int* __restrict__ slot_token)
{
    const int b = blockIdx.x, lane = threadIdx.x;
    int off[NUM_E];
#pragma unroll
    for (int e = 0; e < NUM_E; ++e) off[e] = blockStart[b * NUM_E + e];
    for (int j = 0; j < HCHUNK / 32; ++j) {
        int i = b * HCHUNK + j * 32 + lane;
        int eid = exp_ids[i];
        int rank = 0;
#pragma unroll
        for (int e = 0; e < NUM_E; ++e) {
            unsigned m = (unsigned)__ballot(eid == e);   // wave32: low 32 bits
            if (eid == e) rank = off[e] + __popc(m & ((1u << lane) - 1u));
            off[e] += __popc(m);
        }
        int lpos = rank - expertStart[eid];
        int slot = (lpos < CAP) ? (eid * CAP + lpos) : DUMMYS;
        slot_of_i[i] = slot;
        if (slot != DUMMYS) slot_token[slot] = i >> 1;    // x_exp row i = x[i/K]
    }
}

// Fragment layouts (CDNA5 ISA):
// A 16x32 bf16: lane L holds row M=L&15; elems 0..7 -> K=(L>>4)*8+j ; 8..15 -> K=16+(L>>4)*8+(j-8)
// B 32x16 bf16: lane L holds col N=L&15; code uses elems j -> K=(L>>4)*16 + j, same for A (16 contiguous K)
// C/D f32 16x16: lane L col N=L&15; VGPR r -> row M=r+8*(L>>4)

// ---------------- pass 1: H = silu(x@fc1^T) * (x@gating^T), bf16 out ----------------
__global__ __launch_bounds__(256) void ffn1_kernel(
    const __bf16* __restrict__ xbf, const int* __restrict__ slot_token,
    const __bf16* __restrict__ w1bf, const __bf16* __restrict__ wgbf,
    __bf16* __restrict__ Hbf)
{
    __shared__ __bf16 As[2][64][KP];
    __shared__ __bf16 B1s[2][128][KP];
    __shared__ __bf16 Bgs[2][128][KP];
    __shared__ int tok[64];
    const int e  = blockIdx.z;
    const int m0 = blockIdx.y * 64;
    const int n0 = blockIdx.x * 128;
    const int t  = threadIdx.x;
    if (t < 64) tok[t] = slot_token[e * CAP + m0 + t];
    __syncthreads();

    const int lane = t & 31, wave = t >> 5;
    const int wm = (wave >> 2) * 32;     // 0 / 32
    const int wn = (wave & 3) * 32;      // 0..96

    v8f acc1[2][2], accg[2][2];
#pragma unroll
    for (int a = 0; a < 2; ++a)
#pragma unroll
        for (int bb = 0; bb < 2; ++bb)
#pragma unroll
            for (int r = 0; r < 8; ++r) { acc1[a][bb][r] = 0.f; accg[a][bb][r] = 0.f; }

    auto load_stage = [&](int kb, int buf) {
        { // A: 64 rows x 32 K, gathered via slot_token (manual: needs zero-fill)
            int row = t >> 2, c = (t & 3) * 8;
            int tk = tok[row];
            uint4 val = make_uint4(0u, 0u, 0u, 0u);
            if (tk >= 0) val = *reinterpret_cast<const uint4*>(xbf + (size_t)tk * IND + kb + c);
            *reinterpret_cast<uint4*>(&As[buf][row][c]) = val;
        }
        { // B1/Bg: 128 rows x 32 K -- async DMA straight into LDS when available
            int row = t >> 1, c = (t & 1) * 16;
            const __bf16* p1 = w1bf + ((size_t)(e * HDIM + n0 + row)) * IND + kb + c;
            const __bf16* pg = wgbf + ((size_t)(e * HDIM + n0 + row)) * IND + kb + c;
#if HAVE_ASYNC_LDS
            async_g2l_b128(p1,     &B1s[buf][row][c]);
            async_g2l_b128(p1 + 8, &B1s[buf][row][c + 8]);
            async_g2l_b128(pg,     &Bgs[buf][row][c]);
            async_g2l_b128(pg + 8, &Bgs[buf][row][c + 8]);
#else
            uint4 a0 = *reinterpret_cast<const uint4*>(p1);
            uint4 a1 = *reinterpret_cast<const uint4*>(p1 + 8);
            uint4 g0 = *reinterpret_cast<const uint4*>(pg);
            uint4 g1 = *reinterpret_cast<const uint4*>(pg + 8);
            *reinterpret_cast<uint4*>(&B1s[buf][row][c])     = a0;
            *reinterpret_cast<uint4*>(&B1s[buf][row][c + 8]) = a1;
            *reinterpret_cast<uint4*>(&Bgs[buf][row][c])     = g0;
            *reinterpret_cast<uint4*>(&Bgs[buf][row][c + 8]) = g1;
#endif
        }
    };

    load_stage(0, 0);
    WAIT_ASYNC0();
    __syncthreads();
    const int ar = lane & 15, ak = (lane >> 4) * 16;
    const int bn = lane & 15, bk = (lane >> 4) * 16;

    for (int kb = 0; kb < IND; kb += 32) {
        int cur = (kb >> 5) & 1;
        if (kb + 32 < IND) load_stage(kb + 32, cur ^ 1);
        Frag16 a[2], b1[2], bg[2];
#pragma unroll
        for (int mi = 0; mi < 2; ++mi) {
            a[mi].q[0] = *reinterpret_cast<const uint4*>(&As[cur][wm + mi * 16 + ar][ak]);
            a[mi].q[1] = *reinterpret_cast<const uint4*>(&As[cur][wm + mi * 16 + ar][ak + 8]);
        }
#pragma unroll
        for (int ni = 0; ni < 2; ++ni) {
            b1[ni].q[0] = *reinterpret_cast<const uint4*>(&B1s[cur][wn + ni * 16 + bn][bk]);
            b1[ni].q[1] = *reinterpret_cast<const uint4*>(&B1s[cur][wn + ni * 16 + bn][bk + 8]);
            bg[ni].q[0] = *reinterpret_cast<const uint4*>(&Bgs[cur][wn + ni * 16 + bn][bk]);
            bg[ni].q[1] = *reinterpret_cast<const uint4*>(&Bgs[cur][wn + ni * 16 + bn][bk + 8]);
        }
#pragma unroll
        for (int mi = 0; mi < 2; ++mi)
#pragma unroll
            for (int ni = 0; ni < 2; ++ni) {
                acc1[mi][ni] = __builtin_amdgcn_wmma_f32_16x16x32_bf16(
                    false, a[mi].v, false, b1[ni].v, (short)0, acc1[mi][ni], false, false);
                accg[mi][ni] = __builtin_amdgcn_wmma_f32_16x16x32_bf16(
                    false, a[mi].v, false, bg[ni].v, (short)0, accg[mi][ni], false, false);
            }
        WAIT_ASYNC0();
        __syncthreads();
    }

    // epilogue: H = silu(a)*g   (v_rcp_f32 instead of IEEE divide)
#pragma unroll
    for (int mi = 0; mi < 2; ++mi)
#pragma unroll
        for (int ni = 0; ni < 2; ++ni) {
            int n  = n0 + wn + ni * 16 + (lane & 15);
            int mb = m0 + wm + mi * 16 + ((lane >> 4) * 8);
#pragma unroll
            for (int r = 0; r < 8; ++r) {
                float av = acc1[mi][ni][r];
                float gv = accg[mi][ni][r];
                float sig = __builtin_amdgcn_rcpf(1.f + __expf(-av));
                float h  = av * sig * gv;
                Hbf[((size_t)e * CAP + mb + r) * HDIM + n] = (__bf16)h;
            }
        }
}

// ---------------- pass 2: out = H @ fc2^T (contract over HD) ----------------
__global__ __launch_bounds__(256) void ffn2_kernel(
    const __bf16* __restrict__ Hbf, const __bf16* __restrict__ w2bf,
    float* __restrict__ buf_out)
{
    __shared__ __bf16 As[2][64][KP];
    __shared__ __bf16 Bs[2][128][KP];
    const int e  = blockIdx.z;
    const int m0 = blockIdx.y * 64;
    const int n0 = blockIdx.x * 128;
    const int t  = threadIdx.x;
    const int lane = t & 31, wave = t >> 5;
    const int wm = (wave >> 2) * 32;
    const int wn = (wave & 3) * 32;

    v8f acc[2][2];
#pragma unroll
    for (int a = 0; a < 2; ++a)
#pragma unroll
        for (int bb = 0; bb < 2; ++bb)
#pragma unroll
            for (int r = 0; r < 8; ++r) acc[a][bb][r] = 0.f;

    auto load_stage = [&](int kb, int buf) {
        int rowA = t >> 2, cA = (t & 3) * 8;
        const __bf16* pa = Hbf + ((size_t)e * CAP + m0 + rowA) * HDIM + kb + cA;
        int rowB = t >> 1, cB = (t & 1) * 16;
        const __bf16* pb = w2bf + ((size_t)(e * IND + n0 + rowB)) * HDIM + kb + cB;
#if HAVE_ASYNC_LDS
        async_g2l_b128(pa,     &As[buf][rowA][cA]);
        async_g2l_b128(pb,     &Bs[buf][rowB][cB]);
        async_g2l_b128(pb + 8, &Bs[buf][rowB][cB + 8]);
#else
        uint4 va = *reinterpret_cast<const uint4*>(pa);
        *reinterpret_cast<uint4*>(&As[buf][rowA][cA]) = va;
        uint4 b0 = *reinterpret_cast<const uint4*>(pb);
        uint4 b1 = *reinterpret_cast<const uint4*>(pb + 8);
        *reinterpret_cast<uint4*>(&Bs[buf][rowB][cB])     = b0;
        *reinterpret_cast<uint4*>(&Bs[buf][rowB][cB + 8]) = b1;
#endif
    };

    load_stage(0, 0);
    WAIT_ASYNC0();
    __syncthreads();
    const int ar = lane & 15, ak = (lane >> 4) * 16;
    const int bn = lane & 15, bk = (lane >> 4) * 16;

    for (int kb = 0; kb < HDIM; kb += 32) {
        int cur = (kb >> 5) & 1;
        if (kb + 32 < HDIM) load_stage(kb + 32, cur ^ 1);
        Frag16 a[2], bf[2];
#pragma unroll
        for (int mi = 0; mi < 2; ++mi) {
            a[mi].q[0] = *reinterpret_cast<const uint4*>(&As[cur][wm + mi * 16 + ar][ak]);
            a[mi].q[1] = *reinterpret_cast<const uint4*>(&As[cur][wm + mi * 16 + ar][ak + 8]);
        }
#pragma unroll
        for (int ni = 0; ni < 2; ++ni) {
            bf[ni].q[0] = *reinterpret_cast<const uint4*>(&Bs[cur][wn + ni * 16 + bn][bk]);
            bf[ni].q[1] = *reinterpret_cast<const uint4*>(&Bs[cur][wn + ni * 16 + bn][bk + 8]);
        }
#pragma unroll
        for (int mi = 0; mi < 2; ++mi)
#pragma unroll
            for (int ni = 0; ni < 2; ++ni)
                acc[mi][ni] = __builtin_amdgcn_wmma_f32_16x16x32_bf16(
                    false, a[mi].v, false, bf[ni].v, (short)0, acc[mi][ni], false, false);
        WAIT_ASYNC0();
        __syncthreads();
    }

#pragma unroll
    for (int mi = 0; mi < 2; ++mi)
#pragma unroll
        for (int ni = 0; ni < 2; ++ni) {
            int n  = n0 + wn + ni * 16 + (lane & 15);
            int mb = m0 + wm + mi * 16 + ((lane >> 4) * 8);
#pragma unroll
            for (int r = 0; r < 8; ++r)
                buf_out[((size_t)e * CAP + mb + r) * IND + n] = acc[mi][ni][r];
        }
}

// ---------------- deterministic gather-combine ----------------
__global__ __launch_bounds__(256) void combine_kernel(
    const float* __restrict__ buf_out, const int* __restrict__ slot_of_i,
    const float* __restrict__ wts, float* __restrict__ out)
{
    int idx = blockIdx.x * 256 + threadIdx.x;  // one float4 of out
    int n  = idx / (IND / 4);
    int dc = (idx % (IND / 4)) * 4;
    int s0 = slot_of_i[n * 2], s1 = slot_of_i[n * 2 + 1];
    float w0 = wts[n * 2], w1 = wts[n * 2 + 1];
    if (s0 >= NSLOT) { s0 = 0; w0 = 0.f; }
    if (s1 >= NSLOT) { s1 = 0; w1 = 0.f; }
    float4 a = *reinterpret_cast<const float4*>(buf_out + (size_t)s0 * IND + dc);
    float4 b = *reinterpret_cast<const float4*>(buf_out + (size_t)s1 * IND + dc);
    float4 r;
    r.x = w0 * a.x + w1 * b.x;
    r.y = w0 * a.y + w1 * b.y;
    r.z = w0 * a.z + w1 * b.z;
    r.w = w0 * a.w + w1 * b.w;
    *reinterpret_cast<float4*>(out + (size_t)n * IND + dc) = r;
}

// ---------------- balancing loss: fixed-order reductions ----------------
__global__ __launch_bounds__(256) void preduce_kernel(const float* __restrict__ probs,
                                                      float* __restrict__ partial)
{
    __shared__ float sh[256];
    const int b = blockIdx.x, t = threadIdx.x;
    const int e = t & 7, tl = t >> 3;
    float s = 0.f;
    for (int j = 0; j < 8; ++j) {
        int n = b * 256 + tl + j * 32;
        s += probs[(size_t)n * NUM_E + e];
    }
    sh[t] = s; __syncthreads();
    for (int st = 128; st >= 8; st >>= 1) {
        if (t < st) sh[t] += sh[t + st];
        __syncthreads();
    }
    if (t < 8) partial[b * NUM_E + t] = sh[t];
}

__global__ void finalize_kernel(const float* __restrict__ partial,
                                const int* __restrict__ expertCount,
                                float* __restrict__ loss_out)
{
    if (threadIdx.x != 0 || blockIdx.x != 0) return;
    float loss = 0.f;
    for (int e = 0; e < NUM_E; ++e) {
        float p = 0.f;
        for (int b = 0; b < HBLK; ++b) p += partial[b * NUM_E + e];
        p /= (float)NTOK;
        float f = (float)expertCount[e] / (float)NKTOT;
        loss += p * f;
    }
    loss_out[0] = (float)NUM_E * loss;
}

extern "C" void kernel_launch(void* const* d_in, const int* in_sizes, int n_in,
                              void* d_out, int out_size, void* d_ws, size_t ws_size,
                              hipStream_t stream)
{
    (void)in_sizes; (void)n_in; (void)out_size; (void)ws_size;
    const float* x        = (const float*)d_in[0];
    const float* gate_w   = (const float*)d_in[1];
    const float* fc1_w    = (const float*)d_in[2];
    const float* gating_w = (const float*)d_in[3];
    const float* fc2_w    = (const float*)d_in[4];
    float* out = (float*)d_out;

    char* ws = (char*)d_ws;
    size_t off = 0;
    auto alloc = [&](size_t bytes) -> void* {
        void* p = ws + off;
        off = (off + bytes + 255) & ~(size_t)255;
        return p;
    };
    __bf16* xbf        = (__bf16*)alloc((size_t)NTOK * IND * 2);
    __bf16* w1bf       = (__bf16*)alloc((size_t)NUM_E * HDIM * IND * 2);
    __bf16* wgbf       = (__bf16*)alloc((size_t)NUM_E * HDIM * IND * 2);
    __bf16* w2bf       = (__bf16*)alloc((size_t)NUM_E * IND * HDIM * 2);
    __bf16* Hbf        = (__bf16*)alloc((size_t)NSLOT * HDIM * 2);
    float*  bout       = (float*) alloc((size_t)NSLOT * IND * 4);
    int*    exp_ids    = (int*)   alloc((size_t)NKTOT * 4);
    float*  wts        = (float*) alloc((size_t)NKTOT * 4);
    int*    slot_of_i  = (int*)   alloc((size_t)NKTOT * 4);
    int*    slot_token = (int*)   alloc((size_t)NSLOT * 4);
    float*  probs      = (float*) alloc((size_t)NTOK * NUM_E * 4);
    int*    blockCounts= (int*)   alloc((size_t)HBLK * NUM_E * 4);
    int*    blockStart = (int*)   alloc((size_t)HBLK * NUM_E * 4);
    int*    expertStart= (int*)   alloc((size_t)NUM_E * 4);
    int*    expertCount= (int*)   alloc((size_t)NUM_E * 4);
    float*  partial    = (float*) alloc((size_t)HBLK * NUM_E * 4);

    int n4 = NTOK * IND / 4;
    cast_bf16_kernel<<<n4 / 256, 256, 0, stream>>>(x, xbf, n4);
    n4 = NUM_E * HDIM * IND / 4;
    cast_bf16_kernel<<<n4 / 256, 256, 0, stream>>>(fc1_w,    w1bf, n4);
    cast_bf16_kernel<<<n4 / 256, 256, 0, stream>>>(gating_w, wgbf, n4);
    cast_bf16_kernel<<<n4 / 256, 256, 0, stream>>>(fc2_w,    w2bf, n4);

    init_slots_kernel<<<(NSLOT + 255) / 256, 256, 0, stream>>>(slot_token);
    routing_kernel<<<NTOK / 8, 256, 0, stream>>>(x, gate_w, exp_ids, wts, probs);
    hist_kernel<<<HBLK, 32, 0, stream>>>(exp_ids, blockCounts);
    scan_kernel<<<1, 1, 0, stream>>>(blockCounts, blockStart, expertStart, expertCount);
    scatter_kernel<<<HBLK, 32, 0, stream>>>(exp_ids, blockStart, expertStart, slot_of_i, slot_token);

    dim3 g1(HDIM / 128, CAP / 64, NUM_E);
    ffn1_kernel<<<g1, 256, 0, stream>>>(xbf, slot_token, w1bf, wgbf, Hbf);
    dim3 g2(IND / 128, CAP / 64, NUM_E);
    ffn2_kernel<<<g2, 256, 0, stream>>>(Hbf, w2bf, bout);

    combine_kernel<<<(NTOK * IND / 4) / 256, 256, 0, stream>>>(bout, slot_of_i, wts, out);
    preduce_kernel<<<HBLK, 256, 0, stream>>>(probs, partial);
    finalize_kernel<<<1, 1, 0, stream>>>(partial, expertCount, out + (size_t)NTOK * IND);
}
